// CrossViewSwapAttention_41601053229132
// MI455X (gfx1250) — matrix operand order repaired and checked by reading, not compiled, not run on hardware
//
#include <hip/hip_runtime.h>

// ============================================================================
// CrossViewSwapAttention for MI455X (gfx1250, wave32, WMMA).
//
// All GEMM contractions run on v_wmma_f32_16x16x32_f16 (f16 operands, f32
// accumulation). Elementwise stages (LN, softmax, L2-normalize, BN+ReLU,
// GELU) are fused VALU epilogues around the WMMA tiles with LDS staging.
//
// INPUT ORDER ASSUMPTION: d_in[0..3] = x, feature, I_inv, E_inv; then the
// params pytree flattened with dict keys sorted (JAX tree order):
//   attn1{bk,bq,bv,lnk_b,lnk_g,lnq_b,lnq_g,lnv_b,lnv_g,pb,pw,wk,wq,wv} -> 4..17
//   attn2 same                                                        -> 18..31
//   bev_b=32, bev_w=33, cam_w=34, fl{b,g,m,v,w}=35..39, fp{..}=40..44,
//   img_w=45, mlp1{b1,b2,nb,ng,w1,w2}=46..51, mlp2=52..57,
//   post_b=58, post_g=59
// ============================================================================

typedef __attribute__((ext_vector_type(16))) _Float16 v16h;
typedef __attribute__((ext_vector_type(8)))  float    v8f;

#define NCAM 6
#define DIM  128
#define FW_  80
#define FHW  6400     // 80*80
#define WW   160
#define HW   25600    // 160*160
#define NWIN 400      // 20*20 windows
#define MQ   384      // 6*8*8 query tokens per window
#define MK   96       // 6*4*4 key tokens per window
#define HD_  128      // heads*dh

// ---------------- WMMA fragment helpers (CDNA5 documented layouts) ----------

// A fragment: 16x32 f16, src is M x K row-major.
// lanes 0-15: M=lane, K pairs in VGPRs 0..3 = K 0..7, 4..7 = K 16..23;
// lanes 16-31: same M, K offset +8 in each half (ISA 7.12.2).
__device__ __forceinline__ v16h frag_a(const _Float16* src, int ld) {
  int lane = threadIdx.x & 31;
  int hv = lane >> 4, m = lane & 15;
  v16h f;
#pragma unroll
  for (int j = 0; j < 8; ++j) {
    int k = ((j < 4) ? 0 : 16) + hv * 8 + (j & 3) * 2;
    f[2*j]   = src[m*ld + k];
    f[2*j+1] = src[m*ld + k + 1];
  }
  return f;
}

// B fragment: 32x16 f16, src is K x N row-major.
// N=lane&15; K pairs same as A: VGPRs 0..3 = K 0..7, 4..7 = K 16..23, +8 for lanes 16-31.
__device__ __forceinline__ v16h frag_b(const _Float16* src, int ld) {
  int lane = threadIdx.x & 31;
  int hv = lane >> 4, n = lane & 15;
  v16h f;
#pragma unroll
  for (int j = 0; j < 8; ++j) {
    int k = ((j < 4) ? 0 : 16) + hv * 8 + (j & 3) * 2;
    f[2*j]   = src[k*ld + n];
    f[2*j+1] = src[(k + 1)*ld + n];
  }
  return f;
}

__device__ __forceinline__ v8f wmma32(v16h a, v16h b, v8f c) {
  return __builtin_amdgcn_wmma_f32_16x16x32_f16(false, a, false, b, (short)0, c,
                                                false, false);
}

// C layout: lane&15 = N, (r + 8*(lane>>4)) = M  (per ISA 7.12.2).

// ---------------- setup: c_embed, camera mats, BN fold ---------------------

__global__ void k_setup(const float* Iinv, const float* Einv, const float* camW,
                        const float* fpB, const float* fpG, const float* fpM, const float* fpV,
                        const float* flB, const float* flG, const float* flM, const float* flV,
                        float* cEmb, float* Tm, float* tv,
                        float* bnsFp, float* bnbFp, float* bnsFl, float* bnbFl) {
  int t = threadIdx.x;
  for (int i = t; i < NCAM * DIM; i += 256) {
    int n = i >> 7, o = i & 127;
    float s = 0.f;
    for (int c = 0; c < 4; ++c) s += camW[o*4+c] * Einv[n*16 + c*4 + 3];
    cEmb[i] = s;
  }
  if (t < 72) {  // T[n] = E_inv[:, :3] @ I_inv  (4x3)
    int n = t/12, r = (t%12)/3, c = t%3;
    float s = 0.f;
    for (int j = 0; j < 3; ++j) s += Einv[n*16 + r*4 + j] * Iinv[n*9 + j*3 + c];
    Tm[t] = s;
  }
  if (t < 24) { int n = t/4, r = t%4; tv[t] = Einv[n*16 + r*4 + 3]; }
  if (t < 128) {
    float s = fpG[t] * rsqrtf(fpV[t] + 1e-5f);
    bnsFp[t] = s; bnbFp[t] = fpB[t] - fpM[t]*s;
    float s2 = flG[t] * rsqrtf(flV[t] + 1e-5f);
    bnsFl[t] = s2; bnbFl[t] = flB[t] - flM[t]*s2;
  }
}

__global__ void k_cvt(const float* s, _Float16* d, int n) {
  for (int i = blockIdx.x*blockDim.x + threadIdx.x; i < n; i += gridDim.x*blockDim.x)
    d[i] = (_Float16)s[i];
}

// ---------------- image-plane embedding (normalized) -----------------------

__global__ void k_img_embed(const float* imgW, const float* cEmb,
                            const float* Tm, const float* tv, float* imgEmb) {
  int b = blockIdx.x;                 // cam*6400 + pixel
  int n = b / FHW, p = b % FHW;
  int i = p / FW_, j = p % FW_;
  float px = j * (640.0f / 79.0f), py = i * (640.0f / 79.0f);
  float dv[4];
#pragma unroll
  for (int r = 0; r < 4; ++r)
    dv[r] = Tm[n*12 + r*3 + 0]*px + Tm[n*12 + r*3 + 1]*py + Tm[n*12 + r*3 + 2]
          + tv[n*4 + r];
  int o = threadIdx.x;
  float d = imgW[o*4+0]*dv[0] + imgW[o*4+1]*dv[1] + imgW[o*4+2]*dv[2]
          + imgW[o*4+3]*dv[3] - cEmb[n*DIM + o];
  __shared__ float red[128];
  red[o] = d * d; __syncthreads();
  for (int s = 64; s > 0; s >>= 1) { if (o < s) red[o] += red[o+s]; __syncthreads(); }
  imgEmb[(n*DIM + o)*FHW + p] = d / (sqrtf(red[0]) + 1e-7f);
}

// ---------------- BN + ReLU -> f16 staging ---------------------------------

__global__ void k_feat(const float* feat, const float* bnsFp, const float* bnbFp,
                       const float* bnsFl, const float* bnbFl,
                       _Float16* fFp, _Float16* fFl) {
  int tot = NCAM * DIM * FHW;
  for (int i = blockIdx.x*blockDim.x + threadIdx.x; i < tot; i += gridDim.x*blockDim.x) {
    int c = (i / FHW) % DIM;
    float v = feat[i];
    float a = v*bnsFp[c] + bnbFp[c]; fFp[i] = (_Float16)(a > 0.f ? a : 0.f);
    float b = v*bnsFl[c] + bnbFl[c]; fFl[i] = (_Float16)(b > 0.f ? b : 0.f);
  }
}

// ------ 1x1 convs via WMMA (split kernels -> no per-element divergence) ----

__global__ void k_conv_key(const _Float16* Wm, const _Float16* feats,
                           const float* imgEmb, float* keyB) {
  int tIdx = blockIdx.x * 8 + (threadIdx.x >> 5);   // one 16x16 tile per wave
  int cam  = tIdx / 3200;
  int r2   = tIdx % 3200;
  int mt = r2 / 400, nt = r2 % 400;
  const _Float16* Fm = feats + cam * DIM * FHW;
  v8f acc = {0,0,0,0,0,0,0,0};
#pragma unroll
  for (int kk = 0; kk < 4; ++kk) {
    v16h a = frag_a(Wm + mt*16*DIM + kk*32, DIM);
    v16h b = frag_b(Fm + (kk*32)*FHW + nt*16, FHW);
    acc = wmma32(a, b, acc);
  }
  int lane = threadIdx.x & 31, nn = lane & 15, hv = lane >> 4;
  float e[8];
#pragma unroll
  for (int rr = 0; rr < 8; ++rr)
    e[rr] = imgEmb[(cam*DIM + mt*16 + rr + 8*hv)*FHW + nt*16 + nn];
#pragma unroll
  for (int rr = 0; rr < 8; ++rr)
    keyB[(cam*DIM + mt*16 + rr + 8*hv)*FHW + nt*16 + nn] = acc[rr] + e[rr];
}

__global__ void k_conv_val(const _Float16* Wm, const _Float16* feats, float* valB) {
  int tIdx = blockIdx.x * 8 + (threadIdx.x >> 5);
  int cam  = tIdx / 3200;
  int r2   = tIdx % 3200;
  int mt = r2 / 400, nt = r2 % 400;
  const _Float16* Fm = feats + cam * DIM * FHW;
  v8f acc = {0,0,0,0,0,0,0,0};
#pragma unroll
  for (int kk = 0; kk < 4; ++kk) {
    v16h a = frag_a(Wm + mt*16*DIM + kk*32, DIM);
    v16h b = frag_b(Fm + (kk*32)*FHW + nt*16, FHW);
    acc = wmma32(a, b, acc);
  }
  int lane = threadIdx.x & 31, nn = lane & 15, hv = lane >> 4;
#pragma unroll
  for (int rr = 0; rr < 8; ++rr)
    valB[(cam*DIM + mt*16 + rr + 8*hv)*FHW + nt*16 + nn] = acc[rr];
}

// ---------------- BEV query embedding (normalized) + x ---------------------

__global__ void k_query(const float* bevW, const float* bevB, const float* cEmb,
                        const float* xin, float* query) {
  int b = blockIdx.x;   // cam*25600 + pixel
  int n = b / HW, p = b % HW;
  int h = p / WW, w = p % WW;
  float gx = w * (160.0f / 159.0f), gy = h * (160.0f / 159.0f);
  float g0 = 50.0f - 0.625f * gy;      // Vinv row0 . [gx,gy,1]
  float g1 = 50.0f - 0.625f * gx;      // Vinv row1 . [gx,gy,1]
  int o = threadIdx.x;
  float d = bevW[o*2+0]*g0 + bevW[o*2+1]*g1 + bevB[o] - cEmb[n*DIM + o];
  __shared__ float red[128];
  red[o] = d * d; __syncthreads();
  for (int s = 64; s > 0; s >>= 1) { if (o < s) red[o] += red[o+s]; __syncthreads(); }
  query[(n*DIM + o)*HW + p] = d / (sqrtf(red[0]) + 1e-7f) + xin[o*HW + p];
}

// ---------------- LN + projection for Q ------------------------------------

__global__ void k_lnproj_q(int mode, const float* qsrc, const float* lnG, const float* lnB,
                           const _Float16* Wf, const float* bias, float scale,
                           _Float16* qp) {
  __shared__ float    tokf[64][DIM];
  __shared__ _Float16 tokh[64][DIM];
  __shared__ float mu[64], ri[64];
  int t = threadIdx.x;
  int gbase = blockIdx.x * 64;
  for (int e = t; e < 64*DIM; e += 128) {
    int tt = e >> 7, d = e & 127;
    int g = gbase + tt;
    int Lw = g / MQ, m = g % MQ;
    int n = m >> 6, ij = m & 63, i = ij >> 3, j = ij & 7;
    int wx = Lw / 20, wy = Lw % 20;
    int h = wx*8 + i, w = wy*8 + j;
    float v = (mode == 0) ? qsrc[(n*DIM + d)*HW + h*WW + w]
                          : qsrc[(h*WW + w)*DIM + d];
    tokf[tt][d] = v;
  }
  __syncthreads();
  if (t < 64) {
    float s = 0.f;
    for (int d = 0; d < DIM; ++d) s += tokf[t][d];
    float m_ = s * (1.0f/DIM), vv = 0.f;
    for (int d = 0; d < DIM; ++d) { float q = tokf[t][d] - m_; vv += q*q; }
    mu[t] = m_; ri[t] = rsqrtf(vv*(1.0f/DIM) + 1e-5f);
  }
  __syncthreads();
  for (int e = t; e < 64*DIM; e += 128) {
    int tt = e >> 7, d = e & 127;
    tokh[tt][d] = (_Float16)((tokf[tt][d]-mu[tt])*ri[tt]*lnG[d] + lnB[d]);
  }
  __syncthreads();
  int wv = t >> 5, lane = t & 31, nn = lane & 15, hv = lane >> 4;
  v16h areg[4];
#pragma unroll
  for (int kk = 0; kk < 4; ++kk) areg[kk] = frag_a(&tokh[wv*16][kk*32], DIM);
#pragma unroll
  for (int nt = 0; nt < 8; ++nt) {
    v8f acc = {0,0,0,0,0,0,0,0};
#pragma unroll
    for (int kk = 0; kk < 4; ++kk)
      acc = wmma32(areg[kk], frag_b(Wf + (kk*32)*HD_ + nt*16, HD_), acc);
#pragma unroll
    for (int rr = 0; rr < 8; ++rr) {
      int g2 = gbase + wv*16 + rr + 8*hv;
      qp[g2*HD_ + nt*16 + nn] = (_Float16)((acc[rr] + bias[nt*16+nn]) * scale);
    }
  }
}

// ---------------- LN + projection for K (transposed store) / V -------------

__global__ void k_lnproj_kv(int stage, int transp, const float* src,
                            const float* lnG, const float* lnB,
                            const _Float16* Wf, const float* bias, _Float16* outb) {
  __shared__ float    tokf[64][DIM];
  __shared__ _Float16 tokh[64][DIM];
  __shared__ float mu[64], ri[64];
  int t = threadIdx.x;
  int gbase = blockIdx.x * 64;
  for (int e = t; e < 64*DIM; e += 128) {
    int tt = e >> 7, d = e & 127;
    int g = gbase + tt;
    int Lw = g / MK, m = g % MK;
    int n = m >> 4, r4 = (m >> 2) & 3, c4 = m & 3;
    int wx = Lw / 20, wy = Lw % 20;
    int fh, fw;
    if (stage == 0) { fh = wx*4 + r4;  fw = wy*4 + c4;  }   // block windows
    else            { fh = r4*20 + wx; fw = c4*20 + wy; }   // grid windows
    tokf[tt][d] = src[(n*DIM + d)*FHW + fh*FW_ + fw];
  }
  __syncthreads();
  if (t < 64) {
    float s = 0.f;
    for (int d = 0; d < DIM; ++d) s += tokf[t][d];
    float m_ = s * (1.0f/DIM), vv = 0.f;
    for (int d = 0; d < DIM; ++d) { float q = tokf[t][d] - m_; vv += q*q; }
    mu[t] = m_; ri[t] = rsqrtf(vv*(1.0f/DIM) + 1e-5f);
  }
  __syncthreads();
  for (int e = t; e < 64*DIM; e += 128) {
    int tt = e >> 7, d = e & 127;
    tokh[tt][d] = (_Float16)((tokf[tt][d]-mu[tt])*ri[tt]*lnG[d] + lnB[d]);
  }
  __syncthreads();
  int wv = t >> 5, lane = t & 31, nn = lane & 15, hv = lane >> 4;
  v16h areg[4];
#pragma unroll
  for (int kk = 0; kk < 4; ++kk) areg[kk] = frag_a(&tokh[wv*16][kk*32], DIM);
#pragma unroll
  for (int nt = 0; nt < 8; ++nt) {
    v8f acc = {0,0,0,0,0,0,0,0};
#pragma unroll
    for (int kk = 0; kk < 4; ++kk)
      acc = wmma32(areg[kk], frag_b(Wf + (kk*32)*HD_ + nt*16, HD_), acc);
#pragma unroll
    for (int rr = 0; rr < 8; ++rr) {
      int g2 = gbase + wv*16 + rr + 8*hv;
      int Lw2 = g2 / MK, ml = g2 % MK;
      float o = acc[rr] + bias[nt*16+nn];
      if (transp) outb[(Lw2*HD_ + nt*16 + nn)*MK + ml] = (_Float16)o;  // (d, tok)
      else        outb[(Lw2*MK + ml)*HD_ + nt*16 + nn] = (_Float16)o;  // (tok, d)
    }
  }
}

// ---------------- fused scores -> softmax -> att@V, per (window, head) -----

__global__ void k_attn(const _Float16* qp, const _Float16* kpt,
                       const _Float16* vp, _Float16* obuf) {
  int L = blockIdx.x >> 2, hh = blockIdx.x & 3;
  __shared__ _Float16 kh[32][MK];     // K^T slab: (d, key)
  __shared__ _Float16 vh[MK][32];     // V slab:   (key, d)
  __shared__ float    sc[64][MK];
  __shared__ _Float16 at[64][MK];
  int t = threadIdx.x, wv = t >> 5;
  for (int e = t; e < 32*MK; e += 128) {
    int dd = e / MK, kk = e % MK;
    kh[dd][kk] = kpt[(L*HD_ + hh*32 + dd)*MK + kk];
  }
  for (int e = t; e < MK*32; e += 128) {
    int kk = e / 32, dd = e % 32;
    vh[kk][dd] = vp[(L*MK + kk)*HD_ + hh*32 + dd];
  }
  __syncthreads();
  int lane = t & 31, nn = lane & 15, hv = lane >> 4;
  for (int tb = 0; tb < MQ; tb += 64) {
    v16h a = frag_a(qp + (L*MQ + tb + wv*16)*HD_ + hh*32, HD_);
    if (tb + 64 < MQ)   // overlap next q-chunk fetch with softmax (gfx1250 prefetch)
      __builtin_prefetch(qp + (L*MQ + tb + 64 + wv*16)*HD_ + hh*32, 0, 1);
#pragma unroll
    for (int nt = 0; nt < 6; ++nt) {
      v8f acc = {0,0,0,0,0,0,0,0};
      acc = wmma32(a, frag_b(&kh[0][nt*16], MK), acc);
#pragma unroll
      for (int rr = 0; rr < 8; ++rr)
        sc[wv*16 + rr + 8*hv][nt*16 + nn] = acc[rr];
    }
    __syncthreads();
    if (t < 64) {
      float mx = -1e30f;
      for (int k = 0; k < MK; ++k) mx = fmaxf(mx, sc[t][k]);
      float s = 0.f;
      for (int k = 0; k < MK; ++k) { float e2 = expf(sc[t][k]-mx); sc[t][k] = e2; s += e2; }
      float inv = 1.f / s;
      for (int k = 0; k < MK; ++k) at[t][k] = (_Float16)(sc[t][k]*inv);
    }
    __syncthreads();
    v16h a0 = frag_a(&at[wv*16][0],  MK);
    v16h a1 = frag_a(&at[wv*16][32], MK);
    v16h a2 = frag_a(&at[wv*16][64], MK);
#pragma unroll
    for (int nt = 0; nt < 2; ++nt) {
      v8f acc = {0,0,0,0,0,0,0,0};
      acc = wmma32(a0, frag_b(&vh[0][nt*16],  32), acc);
      acc = wmma32(a1, frag_b(&vh[32][nt*16], 32), acc);
      acc = wmma32(a2, frag_b(&vh[64][nt*16], 32), acc);
#pragma unroll
      for (int rr = 0; rr < 8; ++rr) {
        int m = rr + 8*hv;
        obuf[(L*MQ + tb + wv*16 + m)*HD_ + hh*32 + nt*16 + nn] = (_Float16)acc[rr];
      }
    }
    __syncthreads();
  }
}

// ---------------- output proj + mean over cameras + skip -------------------

__global__ void k_proj(int stage, const _Float16* obuf, const _Float16* pwF,
                       const float* pb, const float* xin, const float* skipB,
                       float* outA) {
  int L = blockIdx.x;
  __shared__ float accs[64][DIM];
  int t = threadIdx.x, wv = t >> 5;
  for (int e = t; e < 64*DIM; e += 128) (&accs[0][0])[e] = 0.f;
  __syncthreads();
  int lane = t & 31, nn = lane & 15, hv = lane >> 4;
  for (int n = 0; n < NCAM; ++n) {
    v16h areg[4];
#pragma unroll
    for (int kk = 0; kk < 4; ++kk)
      areg[kk] = frag_a(obuf + (L*MQ + n*64 + wv*16)*HD_ + kk*32, HD_);
    if (n + 1 < NCAM)
      __builtin_prefetch(obuf + (L*MQ + (n+1)*64 + wv*16)*HD_, 0, 1);
#pragma unroll
    for (int nt = 0; nt < 8; ++nt) {
      v8f acc = {0,0,0,0,0,0,0,0};
#pragma unroll
      for (int kk = 0; kk < 4; ++kk)
        acc = wmma32(areg[kk], frag_b(pwF + (kk*32)*DIM + nt*16, DIM), acc);
#pragma unroll
      for (int rr = 0; rr < 8; ++rr)
        accs[wv*16 + rr + 8*hv][nt*16 + nn] += acc[rr];   // same wave, same rows
    }
  }
  __syncthreads();
  int wx = L / 20, wy = L % 20;
  for (int e = t; e < 64*DIM; e += 128) {
    int ij = e >> 7, d = e & 127;
    int i = ij >> 3, j = ij & 7;
    int p = (wx*8 + i)*WW + wy*8 + j;
    float skip = (stage == 0) ? xin[d*HW + p] : skipB[p*DIM + d];
    outA[p*DIM + d] = accs[ij][d]*(1.0f/6.0f) + pb[d] + skip;
  }
}

// ---------------- MLP block: t + gelu(LN(t)@w1+b1)@w2+b2 -------------------

__global__ void k_mlp(const float* inb, const float* ng, const float* nb,
                      const _Float16* w1F, const float* b1,
                      const _Float16* w2F, const float* b2, float* outb) {
  __shared__ _Float16 th[64][DIM];
  __shared__ _Float16 h1[64][256];
  __shared__ float mu[64], ri[64];
  int t = threadIdx.x;
  int base = blockIdx.x * 64;
  if (t < 64) {
    const float* row = inb + (size_t)(base + t)*DIM;
    float s = 0.f;
    for (int d = 0; d < DIM; ++d) s += row[d];
    float m_ = s * (1.0f/DIM), vv = 0.f;
    for (int d = 0; d < DIM; ++d) { float q = row[d] - m_; vv += q*q; }
    mu[t] = m_; ri[t] = rsqrtf(vv*(1.0f/DIM) + 1e-5f);
  }
  __syncthreads();
  for (int e = t; e < 64*DIM; e += 128) {
    int tt = e >> 7, d = e & 127;
    float v = inb[(size_t)(base + tt)*DIM + d];
    th[tt][d] = (_Float16)((v - mu[tt])*ri[tt]*ng[d] + nb[d]);
  }
  __syncthreads();
  int wv = t >> 5, lane = t & 31, nn = lane & 15, hv = lane >> 4;
  {
    v16h areg[4];
#pragma unroll
    for (int kk = 0; kk < 4; ++kk) areg[kk] = frag_a(&th[wv*16][kk*32], DIM);
#pragma unroll
    for (int nt = 0; nt < 16; ++nt) {
      v8f acc = {0,0,0,0,0,0,0,0};
#pragma unroll
      for (int kk = 0; kk < 4; ++kk)
        acc = wmma32(areg[kk], frag_b(w1F + (kk*32)*256 + nt*16, 256), acc);
#pragma unroll
      for (int rr = 0; rr < 8; ++rr) {
        float u = acc[rr] + b1[nt*16+nn];
        float g = 0.5f*u*(1.0f + erff(u*0.70710678f));   // exact GELU
        h1[wv*16 + rr + 8*hv][nt*16 + nn] = (_Float16)g;
      }
    }
  }
  __syncthreads();
  {
    v16h areg[8];
#pragma unroll
    for (int kk = 0; kk < 8; ++kk) areg[kk] = frag_a(&h1[wv*16][kk*32], 256);
#pragma unroll
    for (int nt = 0; nt < 8; ++nt) {
      v8f acc = {0,0,0,0,0,0,0,0};
#pragma unroll
      for (int kk = 0; kk < 8; ++kk)
        acc = wmma32(areg[kk], frag_b(w2F + (kk*32)*DIM + nt*16, DIM), acc);
#pragma unroll
      for (int rr = 0; rr < 8; ++rr) {
        int p = base + wv*16 + rr + 8*hv;
        int d = nt*16 + nn;
        outb[(size_t)p*DIM + d] = inb[(size_t)p*DIM + d] + acc[rr] + b2[d];
      }
    }
  }
}

// ---------------- post LayerNorm + transpose to (D,H,W) --------------------

__global__ void k_post(const float* inb, const float* pg, const float* pbb, float* outp) {
  __shared__ float mu[64], ri[64];
  int t = threadIdx.x;
  int base = blockIdx.x * 64;
  if (t < 64) {
    const float* row = inb + (size_t)(base + t)*DIM;
    float s = 0.f;
    for (int d = 0; d < DIM; ++d) s += row[d];
    float m_ = s * (1.0f/DIM), vv = 0.f;
    for (int d = 0; d < DIM; ++d) { float q = row[d] - m_; vv += q*q; }
    mu[t] = m_; ri[t] = rsqrtf(vv*(1.0f/DIM) + 1e-5f);
  }
  __syncthreads();
  for (int e = t; e < 64*DIM; e += 128) {
    int tt = e >> 7, d = e & 127;
    int p = base + tt;
    float v = inb[(size_t)p*DIM + d];
    outp[(size_t)d*HW + p] = (v - mu[tt])*ri[tt]*pg[d] + pbb[d];
  }
}

// ============================================================================
// Host launch
// ============================================================================

extern "C" void kernel_launch(void* const* d_in, const int* in_sizes, int n_in,
                              void* d_out, int out_size, void* d_ws, size_t ws_size,
                              hipStream_t stream) {
  (void)in_sizes; (void)out_size; (void)ws_size;
  const float* P[64];
  for (int i = 0; i < n_in && i < 64; ++i) P[i] = (const float*)d_in[i];
  const float* xin  = P[0];
  const float* feat = P[1];
  const float* Iinv = P[2];
  const float* Einv = P[3];
  // param leaf indices (see header comment)
  const int A1 = 4, A2 = 18;
  const int O_BK=0, O_BQ=1, O_BV=2, O_LNKB=3, O_LNKG=4, O_LNQB=5, O_LNQG=6,
            O_LNVB=7, O_LNVG=8, O_PB=9, O_PW=10, O_WK=11, O_WQ=12, O_WV=13;
  const int I_BEVB=32, I_BEVW=33, I_CAMW=34;
  const int FL=35, FP=40;               // {b,g,m,v,w}
  const int I_IMGW=45;
  const int M1=46, M2=52;               // {b1,b2,nb,ng,w1,w2}
  const int O_B1=0, O_B2=1, O_NB=2, O_NG=3, O_W1=4, O_W2=5;
  const int I_POSTB=58, I_POSTG=59;

  // ---- workspace bump allocator ----
  char* wsb = (char*)d_ws;
  size_t off = 0;
  auto alloc = [&](size_t bytes) -> void* {
    void* p = wsb + off;
    off = (off + bytes + 255) & ~(size_t)255;
    return p;
  };
  float* cEmb   = (float*)alloc(NCAM*DIM*4);
  float* Tm     = (float*)alloc(72*4);
  float* tv     = (float*)alloc(24*4);
  float* bnsFp  = (float*)alloc(DIM*4);
  float* bnbFp  = (float*)alloc(DIM*4);
  float* bnsFl  = (float*)alloc(DIM*4);
  float* bnbFl  = (float*)alloc(DIM*4);
  float* imgEmb = (float*)alloc((size_t)NCAM*DIM*FHW*4);
  float* keyB   = (float*)alloc((size_t)NCAM*DIM*FHW*4);
  float* valB   = (float*)alloc((size_t)NCAM*DIM*FHW*4);
  float* query  = (float*)alloc((size_t)NCAM*DIM*HW*4);
  float* bufA   = (float*)alloc((size_t)HW*DIM*4);
  float* bufB   = (float*)alloc((size_t)HW*DIM*4);
  _Float16* fFp = (_Float16*)alloc((size_t)NCAM*DIM*FHW*2);
  _Float16* fFl = (_Float16*)alloc((size_t)NCAM*DIM*FHW*2);
  _Float16* wqF[2], *wkF[2], *wvF[2], *pwF[2], *w1F[2], *w2F[2];
  for (int s = 0; s < 2; ++s) {
    wqF[s] = (_Float16*)alloc(DIM*HD_*2);
    wkF[s] = (_Float16*)alloc(DIM*HD_*2);
    wvF[s] = (_Float16*)alloc(DIM*HD_*2);
    pwF[s] = (_Float16*)alloc(HD_*DIM*2);
    w1F[s] = (_Float16*)alloc(DIM*256*2);
    w2F[s] = (_Float16*)alloc(256*DIM*2);
  }
  _Float16* fpwF = (_Float16*)alloc(DIM*DIM*2);
  _Float16* flwF = (_Float16*)alloc(DIM*DIM*2);
  _Float16* qp   = (_Float16*)alloc((size_t)NWIN*MQ*HD_*2);
  _Float16* kpt  = (_Float16*)alloc((size_t)NWIN*HD_*MK*2);
  _Float16* vp   = (_Float16*)alloc((size_t)NWIN*MK*HD_*2);
  _Float16* obuf = (_Float16*)alloc((size_t)NWIN*MQ*HD_*2);

  // ---- prep ----
  k_setup<<<1, 256, 0, stream>>>(Iinv, Einv, P[I_CAMW],
                                 P[FP+0], P[FP+1], P[FP+2], P[FP+3],
                                 P[FL+0], P[FL+1], P[FL+2], P[FL+3],
                                 cEmb, Tm, tv, bnsFp, bnbFp, bnsFl, bnbFl);
  auto cvt = [&](int idx, _Float16* dst, int n) {
    k_cvt<<<64, 256, 0, stream>>>(P[idx], dst, n);
  };
  for (int s = 0; s < 2; ++s) {
    int ab = (s == 0) ? A1 : A2, mb = (s == 0) ? M1 : M2;
    cvt(ab + O_WQ, wqF[s], DIM*HD_);
    cvt(ab + O_WK, wkF[s], DIM*HD_);
    cvt(ab + O_WV, wvF[s], DIM*HD_);
    cvt(ab + O_PW, pwF[s], HD_*DIM);
    cvt(mb + O_W1, w1F[s], DIM*256);
    cvt(mb + O_W2, w2F[s], 256*DIM);
  }
  cvt(FP + 4, fpwF, DIM*DIM);
  cvt(FL + 4, flwF, DIM*DIM);

  // ---- embeddings / features / convs / query ----
  k_img_embed<<<NCAM*FHW, 128, 0, stream>>>(P[I_IMGW], cEmb, Tm, tv, imgEmb);
  k_feat<<<4800, 256, 0, stream>>>(feat, bnsFp, bnbFp, bnsFl, bnbFl, fFp, fFl);
  k_conv_key<<<2400, 256, 0, stream>>>(fpwF, fFp, imgEmb, keyB);
  k_conv_val<<<2400, 256, 0, stream>>>(flwF, fFl, valB);
  k_query<<<NCAM*HW, 128, 0, stream>>>(P[I_BEVW], P[I_BEVB], cEmb, xin, query);

  const float qscale = 0.1767766953f;  // 1/sqrt(32)

  // ---- two attention stages ----
  for (int s = 0; s < 2; ++s) {
    int ab = (s == 0) ? A1 : A2, mb = (s == 0) ? M1 : M2;
    const float* qsrc = (s == 0) ? query : bufB;
    k_lnproj_q<<<NWIN*MQ/64, 128, 0, stream>>>(s, qsrc, P[ab+O_LNQG], P[ab+O_LNQB],
                                               wqF[s], P[ab+O_BQ], qscale, qp);
    k_lnproj_kv<<<NWIN*MK/64, 128, 0, stream>>>(s, 1, keyB, P[ab+O_LNKG], P[ab+O_LNKB],
                                                wkF[s], P[ab+O_BK], kpt);
    k_lnproj_kv<<<NWIN*MK/64, 128, 0, stream>>>(s, 0, valB, P[ab+O_LNVG], P[ab+O_LNVB],
                                                wvF[s], P[ab+O_BV], vp);
    k_attn<<<NWIN*4, 128, 0, stream>>>(qp, kpt, vp, obuf);
    k_proj<<<NWIN, 128, 0, stream>>>(s, obuf, pwF[s], P[ab+O_PB], xin, bufB, bufA);
    k_mlp<<<HW/64, 128, 0, stream>>>(bufA, P[mb+O_NG], P[mb+O_NB],
                                     w1F[s], P[mb+O_B1], w2F[s], P[mb+O_B2], bufB);
  }

  // ---- post LN + transpose to (D, H, W) ----
  k_post<<<HW/64, 128, 0, stream>>>(bufB, P[I_POSTG], P[I_POSTB], (float*)d_out);
}
